// MultiGaussianActor_29368986370237
// MI455X (gfx1250) — matrix operand order repaired and checked by reading, not compiled, not run on hardware
//
#include <hip/hip_runtime.h>
#include <cmath>

typedef __attribute__((ext_vector_type(2)))  float  v2f;
typedef __attribute__((ext_vector_type(4)))  float  v4f;
typedef __attribute__((ext_vector_type(8)))  float  v8f;
typedef __attribute__((ext_vector_type(4)))  __bf16 v4bf;
typedef __attribute__((ext_vector_type(8)))  __bf16 v8bf;
typedef __attribute__((ext_vector_type(16))) __bf16 v16bf;

#define B_N 4096
#define D_N 1024
#define A_N 64
#define T_N 16
#define MBLK 32            // rows per expert block (2 WMMA m-tiles)
#define HSTRIDE 1028       // f32 H rows: LDS bank = (4*m + k) % 64, conflict-free

// ---------------- kernel 0: zero the per-task counters ----------------
__global__ void mga_zero_counts(int* __restrict__ counts) {
    if (threadIdx.x < T_N) counts[threadIdx.x] = 0;
}

// ---------------- kernel 1: route rows to experts, emit std ----------------
__global__ void mga_route(const float* __restrict__ actor_index,
                          const float* __restrict__ sigma,
                          float* __restrict__ out_std,
                          int* __restrict__ counts,
                          int* __restrict__ bucket) {
    int b = blockIdx.x * blockDim.x + threadIdx.x;
    if (b >= B_N) return;
    const float* ai = actor_index + (size_t)b * T_N;
    float mx = ai[0];
    int t = 0;
#pragma unroll
    for (int i = 1; i < T_N; ++i) {
        float v = ai[i];
        if (v > mx) { mx = v; t = i; }   // first-max wins == one-hot argmax
    }
    int pos = atomicAdd(&counts[t], 1);
    bucket[t * B_N + pos] = b;

    const float* sg = sigma + (size_t)t * A_N;
    float* os = out_std + (size_t)b * A_N;
#pragma unroll 8
    for (int a = 0; a < A_N; ++a) os[a] = expf(sg[a]);
}

// ------- kernel P: pack W1 into split-bf16 WMMA B-fragment order -------
// layout: frag[t][nt][k32][{hi,lo}][lane][16]  (bf16)
// lane half0 holds K = k32*32 + 0..15 of column n, half1 holds K = +16..31.
__global__ void mga_pack_w1(const float* __restrict__ W1, __bf16* __restrict__ frag) {
    int idx  = blockIdx.x * 256 + threadIdx.x;   // 16*64*32*32 = 1,048,576
    int lane = idx & 31;
    int k32  = (idx >> 5) & 31;
    int nt   = (idx >> 10) & 63;
    int t    = idx >> 16;
    int n    = (nt << 4) + (lane & 15);
    int kb   = (k32 << 5) + ((lane >> 4) << 4);
    const float* w = W1 + ((size_t)t << 20) + n;
    v8bf h0, h1, l0, l1;
#pragma unroll
    for (int j = 0; j < 8; ++j) {
        float f = w[(size_t)(kb + j) << 10];
        __bf16 h = (__bf16)f;
        h0[j] = h; l0[j] = (__bf16)(f - (float)h);
    }
#pragma unroll
    for (int j = 0; j < 8; ++j) {
        float f = w[(size_t)(kb + 8 + j) << 10];
        __bf16 h = (__bf16)f;
        h1[j] = h; l1[j] = (__bf16)(f - (float)h);
    }
    size_t blk = (((size_t)t * 64 + nt) * 32 + k32) << 10;   // 1024 bf16 per k32 block
    __bf16* po = frag + blk + (lane << 4);
    *(v8bf*)(po)       = h0;
    *(v8bf*)(po + 8)   = h1;
    *(v8bf*)(po + 512) = l0;
    *(v8bf*)(po + 520) = l1;
}

// ---- helpers ----
__device__ __forceinline__ v16bf ld_frag(const __bf16* p) {
    v8bf a = *(const v8bf*)p;
    v8bf b = *(const v8bf*)(p + 8);
    return __builtin_shufflevector(a, b, 0,1,2,3,4,5,6,7,8,9,10,11,12,13,14,15);
}

#define WMMA_BF16(A, Bm, C) \
    __builtin_amdgcn_wmma_f32_16x16x32_bf16(false, (A), false, (Bm), (short)0, (C), false, false)

// one k32 step: load 4 A-fragments from LDS (16-bit DS offsets off one base),
// run 6 independent wmma chains (hi*hi, hi*lo, lo*hi) x 2 m-tiles
__device__ __forceinline__ void mma_step(const __bf16* xb, v16bf Bhi, v16bf Blo,
                                         v8f& a0a, v8f& a0b, v8f& a0c,
                                         v8f& a1a, v8f& a1b, v8f& a1c) {
    v16bf Ahi0 = ld_frag(xb);            // [k32][mt=0][hi]
    v16bf Alo0 = ld_frag(xb + 512);      // [k32][mt=0][lo]
    v16bf Ahi1 = ld_frag(xb + 1024);     // [k32][mt=1][hi]
    v16bf Alo1 = ld_frag(xb + 1536);     // [k32][mt=1][lo]
    a0a = WMMA_BF16(Ahi0, Bhi, a0a);
    a1a = WMMA_BF16(Ahi1, Bhi, a1a);
    a0b = WMMA_BF16(Ahi0, Blo, a0b);
    a1b = WMMA_BF16(Ahi1, Blo, a1b);
    a0c = WMMA_BF16(Alo0, Bhi, a0c);
    a1c = WMMA_BF16(Alo1, Bhi, a1c);
}

// forced LDS layout: XF first so A-fragment DS offsets fit 16-bit immediates
struct __align__(16) SmemT {
    __bf16 XF[32 * 2 * 2 * 512];       // [k32][mt][{hi,lo}][lane][16] -> 128 KB @ offset 0
    float  Hlds[MBLK * HSTRIDE];       // f32 H for exact stage 2
    int    rows[MBLK];
};

// ---------------- kernel 2: per-task 32-row expert MLP via WMMA ----------------
__global__ __launch_bounds__(256)
void mga_expert(const float* __restrict__ feature,
                const float* __restrict__ W1, const float* __restrict__ b1,
                const float* __restrict__ W2, const float* __restrict__ b2,
                const int* __restrict__ counts, const int* __restrict__ bucket,
                const __bf16* __restrict__ fragW1,     // packed W1 or nullptr
                float* __restrict__ out_mean) {
    __shared__ SmemT sm;

    const int t         = blockIdx.y;
    const int tileStart = blockIdx.x * MBLK;
    const int cnt       = counts[t];
    if (tileStart >= cnt) return;

    const int tid   = threadIdx.x;
    const int lane  = tid & 31;
    const int wave  = tid >> 5;
    const int m     = lane & 15;            // A-row / B-col / D-col index
    const int khalf = (lane >> 4) * 2;      // K sub-offset (f32 wmma, stage 2)
    const int mhi   = (lane >> 4) * 8;      // C/D row-group offset
    const int khalf16 = (lane >> 4) << 4;   // K sub-offset (bf16 wmma B operand)

    if (tid < MBLK) {
        int g = tileStart + tid;
        sm.rows[tid] = (g < cnt) ? bucket[t * B_N + g] : -1;
    }
    __syncthreads();

    // gather MBLK rows of feature, split into bf16 hi/lo, store as A-fragments
    for (int e = tid; e < MBLK * (D_N / 4); e += 256) {
        int row = e >> 8;        // D_N/4 == 256
        int c4  = e & 255;
        int k   = c4 * 4;
        v4f v = (v4f){0.f, 0.f, 0.f, 0.f};
        int r = sm.rows[row];
        if (r >= 0) v = ((const v4f*)(feature + (size_t)r * D_N))[c4];
        v4bf hi4, lo4;
#pragma unroll
        for (int j = 0; j < 4; ++j) {
            float f = v[j];
            __bf16 h = (__bf16)f;
            hi4[j] = h; lo4[j] = (__bf16)(f - (float)h);
        }
        int mt = row >> 4, mm = row & 15;
        int k32i = k >> 5, kr = k & 31;
        int halfsel = kr >> 4;
        int pos = kr & 15;
        int addr = ((k32i * 2 + mt) * 2) * 512 + (halfsel * 16 + mm) * 16 + pos;
        *(v4bf*)&sm.XF[addr]       = hi4;   // stream 0 = hi
        *(v4bf*)&sm.XF[addr + 512] = lo4;   // stream 1 = lo
    }
    __syncthreads();

    const float* b1t = b1 + (size_t)t * D_N;

    // ---- stage 1: H = relu(X[32x1024] @ W1t[1024x1024] + b1), split-bf16 x3 ----
    for (int nt = wave; nt < D_N / 16; nt += 8) {
        const int nbase = nt * 16;
        float bias = b1t[nbase + m];
        v8f acc0a = (v8f){bias, bias, bias, bias, bias, bias, bias, bias};
        v8f acc1a = acc0a;
        v8f acc0b = (v8f){0,0,0,0,0,0,0,0}, acc0c = acc0b;
        v8f acc1b = acc0b, acc1c = acc0b;
        const __bf16* xb0 = sm.XF + lane * 16;

        if (fragW1) {
            const __bf16* fb = fragW1 + (((size_t)t * 64 + nt) << 15) + (lane << 4);
#pragma unroll 2
            for (int k32 = 0; k32 < D_N / 32; ++k32) {
                const __bf16* pb = fb + (k32 << 10);
                __builtin_prefetch(pb + 4096, 0, 3);             // 4 k32-steps ahead
                v16bf Bhi = ld_frag(pb);
                v16bf Blo = ld_frag(pb + 512);
                mma_step(xb0 + (k32 << 11), Bhi, Blo,
                         acc0a, acc0b, acc0c, acc1a, acc1b, acc1c);
            }
        } else {
            const float* bp = W1 + ((size_t)t << 20) + nbase + m;   // column walk
#pragma unroll 2
            for (int k32 = 0; k32 < D_N / 32; ++k32) {
                const int kb0 = (k32 << 5) + khalf16;
                v16bf Bhi, Blo;
#pragma unroll
                for (int j = 0; j < 16; ++j) {
                    float f = bp[(size_t)(kb0 + j) << 10];
                    __bf16 h = (__bf16)f;
                    Bhi[j] = h; Blo[j] = (__bf16)(f - (float)h);
                }
                mma_step(xb0 + (k32 << 11), Bhi, Blo,
                         acc0a, acc0b, acc0c, acc1a, acc1b, acc1c);
            }
        }
        v8f s0 = acc0a + acc0b + acc0c;
        v8f s1 = acc1a + acc1b + acc1c;
#pragma unroll
        for (int r8 = 0; r8 < 8; ++r8) {
            float v0 = s0[r8] > 0.f ? s0[r8] : 0.f;   // ReLU
            float v1 = s1[r8] > 0.f ? s1[r8] : 0.f;
            sm.Hlds[(r8 + mhi)      * HSTRIDE + nbase + m] = v0;
            sm.Hlds[(r8 + mhi + 16) * HSTRIDE + nbase + m] = v1;
        }
    }
    __syncthreads();

    // ---- stage 2 (exact f32): MU = tanh(H[32x1024] @ W2t[1024x64] + b2) ----
    // 8 waves = 2 m-tiles x 4 n-tiles; even/odd K chains hide wmma latency
    {
        const int mt    = wave >> 2;          // 0 or 1
        const int nbase = (wave & 3) * 16;
        const float* W2t = W2 + (size_t)t * D_N * A_N;
        const float* b2t = b2 + (size_t)t * A_N;
        float bias = b2t[nbase + m];
        v8f accE = (v8f){bias, bias, bias, bias, bias, bias, bias, bias};
        v8f accO = (v8f){0,0,0,0,0,0,0,0};
        const float* bp = W2t + nbase + m;
        const int mrowbase = mt * 16;
        const float* hrow = &sm.Hlds[(m + mrowbase) * HSTRIDE];
#pragma unroll 4
        for (int k = 0; k < D_N; k += 8) {
            const int kk = k + khalf;
            v2f aE = *(const v2f*)(hrow + kk);
            v2f aO = *(const v2f*)(hrow + kk + 4);
            v2f bE, bO;
            bE.x = bp[(size_t)(kk    ) * A_N];
            bE.y = bp[(size_t)(kk + 1) * A_N];
            bO.x = bp[(size_t)(kk + 4) * A_N];
            bO.y = bp[(size_t)(kk + 5) * A_N];
            accE = __builtin_amdgcn_wmma_f32_16x16x4_f32(false, aE, false, bE, (short)0, accE, false, false);
            accO = __builtin_amdgcn_wmma_f32_16x16x4_f32(false, aO, false, bO, (short)0, accO, false, false);
        }
        v8f acc = accE + accO;
#pragma unroll
        for (int r8 = 0; r8 < 8; ++r8) {
            int grow = sm.rows[r8 + mhi + mrowbase];
            if (grow >= 0)
                out_mean[(size_t)grow * A_N + nbase + m] = tanhf(acc[r8]); // ACTION_MAX=1
        }
    }
}

// ---------------- launcher ----------------
extern "C" void kernel_launch(void* const* d_in, const int* in_sizes, int n_in,
                              void* d_out, int out_size, void* d_ws, size_t ws_size,
                              hipStream_t stream) {
    const float* feature     = (const float*)d_in[0];   // [B, D]
    const float* actor_index = (const float*)d_in[1];   // [B, T]
    const float* W1          = (const float*)d_in[2];   // [T, D, D]
    const float* b1          = (const float*)d_in[3];   // [T, D]
    const float* W2          = (const float*)d_in[4];   // [T, D, A]
    const float* b2          = (const float*)d_in[5];   // [T, A]
    const float* sigma       = (const float*)d_in[6];   // [T, A]

    float* out_mean = (float*)d_out;                    // [B, A]
    float* out_std  = out_mean + (size_t)B_N * A_N;     // [B, A]

    int* counts = (int*)d_ws;                 // [T]
    int* bucket = counts + T_N;               // [T, B]
    const size_t frag_off_bytes = ((size_t)(T_N + T_N * B_N) * sizeof(int) + 255) & ~(size_t)255;
    __bf16* fragW1 = (__bf16*)((char*)d_ws + frag_off_bytes);
    const size_t frag_bytes = (size_t)T_N * D_N * D_N * 2 * sizeof(__bf16);  // hi+lo, 64 MB
    const bool use_frag = ws_size >= frag_off_bytes + frag_bytes;

    if (use_frag) {
        mga_pack_w1<<<(T_N * 64 * 32 * 32) / 256, 256, 0, stream>>>(W1, fragW1);
    }
    mga_zero_counts<<<1, 32, 0, stream>>>(counts);
    mga_route<<<B_N / 256, 256, 0, stream>>>(actor_index, sigma, out_std, counts, bucket);
    mga_expert<<<dim3(B_N / MBLK, T_N), 256, 0, stream>>>(
        feature, W1, b1, W2, b2, counts, bucket,
        use_frag ? fragW1 : (const __bf16*)nullptr, out_mean);
}
